// GatedCrossAttention_77017353552534
// MI455X (gfx1250) — matrix operand order repaired and checked by reading, not compiled, not run on hardware
//
#include <hip/hip_runtime.h>
#include <hip/hip_bf16.h>

// ---------------------------------------------------------------------------
// MI455X (gfx1250) implementation of the gated cross-attention block.
// All matrix products run through v_wmma_f32_16x16x32_bf16 (bf16 in, f32 acc):
// on CDNA5 the bf16 WMMA retires K=32 per issue vs K=4 for the f32 WMMA, so
// bf16 is the precision the matrix pipes reward; fp32 accumulation keeps the
// epilogue math (LN / softmax / GELU / gating) in full precision.
//
// V is explicitly transposed per (batch, head) (2 MB total) so that every
// GEMM is the "NT" form: both A and B fragments are contiguous K-major
// loads that match the ISA A(16x32)/B(32x16) bf16 VGPR layouts. Tile bounds
// are handled by clamping (not branching) so the GEMM inner loop is pure
// global_load_b128 + v_wmma with no exec-mask manipulation.
// ---------------------------------------------------------------------------

typedef __attribute__((ext_vector_type(16))) __bf16 v16bf;
typedef __attribute__((ext_vector_type(8)))  float          v8f;
typedef __attribute__((ext_vector_type(8)))  unsigned short v8us;

union FragBF {
  v16bf bf;
  v8us  h[2];
  unsigned short s[16];
};

__device__ __forceinline__ unsigned short f2bf(float f) {
  unsigned int u = __float_as_uint(f);
  u += 0x7FFFu + ((u >> 16) & 1u);   // round-to-nearest-even
  return (unsigned short)(u >> 16);
}

__device__ __forceinline__ float wave_sum(float v) {
#pragma unroll
  for (int o = 16; o; o >>= 1) v += __shfl_xor(v, o, 32);
  return v;
}
__device__ __forceinline__ float wave_max(float v) {
#pragma unroll
  for (int o = 16; o; o >>= 1) v = fmaxf(v, __shfl_xor(v, o, 32));
  return v;
}

// ---------------------------------------------------------------------------
// Batched bf16 GEMM, f32 accumulate:   C[M,N] = A[M,K] * B[N,K]^T
// A: [M,K] row-major (K-contiguous); B: [N,K] row-major (K-contiguous).
// blockIdx.z selects (batch, head): z = zb*nH + zh, independent strides.
// Workgroup = 8 waves; each wave computes a 64x32 macro-tile = 4x2 WMMA tiles.
// Fragment layouts per CDNA5 ISA §7.12.2:
//   A 16x32 bf16: lane(half=0) row r holds K {k0..k0+7, k0+16..k0+23},
//                 lane(half=1) holds K {k0+8..k0+15, k0+24..k0+31}.
//   B 32x16 bf16: lane col r, same K split as A (half h: k0+8h..+7, k0+16+8h..+7).
//   C/D 16x16 f32: lane col r, vgpr v -> row (8*half + v).
// M and N must be multiples of 16 (true for every call site); out-of-range
// tiles are clamped for loads and skipped at store time.
// ---------------------------------------------------------------------------
__global__ __launch_bounds__(256) void gemm_bf16_wmma(
    const unsigned short* __restrict__ A, int lda, long long sAb, long long sAh,
    const unsigned short* __restrict__ Bm, int ldb, long long sBb, long long sBh,
    float* __restrict__ C, int ldc, long long sCb, long long sCh,
    int M, int N, int K, int nH)
{
  const int z  = blockIdx.z;
  const int zb = z / nH, zh = z % nH;
  A  += (long long)zb * sAb + (long long)zh * sAh;
  Bm += (long long)zb * sBb + (long long)zh * sBh;
  C  += (long long)zb * sCb + (long long)zh * sCh;

  const int wave = threadIdx.x >> 5;
  const int lane = threadIdx.x & 31;
  const int r    = lane & 15;
  const int half = lane >> 4;

  const int m_wave = blockIdx.y * 128 + (wave & 1) * 64;  // 2 waves along M
  const int n_wave = blockIdx.x * 128 + (wave >> 1) * 32; // 4 waves along N

  // Clamped fragment base pointers (always in-bounds; no branches in K loop).
  const unsigned short* ap[4];
  const unsigned short* bp[2];
#pragma unroll
  for (int i = 0; i < 4; ++i) {
    int mt = m_wave + i * 16;
    if (mt > M - 16) mt = M - 16;
    ap[i] = A + (long long)(mt + r) * lda + half * 8;
  }
#pragma unroll
  for (int j = 0; j < 2; ++j) {
    int nt = n_wave + j * 16;
    if (nt > N - 16) nt = N - 16;
    bp[j] = Bm + (long long)(nt + r) * ldb + half * 8;
  }

  v8f acc[4][2];
#pragma unroll
  for (int i = 0; i < 4; ++i)
#pragma unroll
    for (int j = 0; j < 2; ++j)
#pragma unroll
      for (int v = 0; v < 8; ++v) acc[i][j][v] = 0.0f;

  for (int k0 = 0; k0 < K; k0 += 32) {
    FragBF a[4], b[2];
#pragma unroll
    for (int i = 0; i < 4; ++i) {
      a[i].h[0] = *(const v8us*)(ap[i] + k0);        // K +0..7   (of this half)
      a[i].h[1] = *(const v8us*)(ap[i] + k0 + 16);   // K +16..23 (of this half)
    }
#pragma unroll
    for (int j = 0; j < 2; ++j) {
      b[j].h[0] = *(const v8us*)(bp[j] + k0);
      b[j].h[1] = *(const v8us*)(bp[j] + k0 + 16);
      // pull the shared B (weight) stream ahead: lowers to global_prefetch_b8
      __builtin_prefetch(bp[j] + k0 + 128, 0, 1);
    }
#pragma unroll
    for (int i = 0; i < 4; ++i)
#pragma unroll
      for (int j = 0; j < 2; ++j)
        acc[i][j] = __builtin_amdgcn_wmma_f32_16x16x32_bf16(
            false, a[i].bf, false, b[j].bf, (short)0, acc[i][j], false, false);
  }

#pragma unroll
  for (int i = 0; i < 4; ++i) {
    const int mt = m_wave + i * 16;
    if (mt >= M) continue;
#pragma unroll
    for (int j = 0; j < 2; ++j) {
      const int nt = n_wave + j * 16;
      if (nt >= N) continue;
      float* cp = C + (long long)(mt + half * 8) * ldc + nt + r;
#pragma unroll
      for (int v = 0; v < 8; ++v) cp[(long long)v * ldc] = acc[i][j][v];
    }
  }
}

// --------------------------- epilogue kernels ------------------------------

// Per-(batch,head) transpose of the V slice:  Vt[z, d, j] = V[b, j, h*DH + d]
// so the attention P*V product becomes a clean NT GEMM. Only 2 MB of traffic.
__global__ void vtrans_kernel(const unsigned short* __restrict__ V, int ldv,
                              long long sVb, int nH, int NKd, int DHd,
                              unsigned short* __restrict__ Vt, long long total)
{
  long long idx = (long long)blockIdx.x * blockDim.x + threadIdx.x;
  const long long stride = (long long)gridDim.x * blockDim.x;
  for (; idx < total; idx += stride) {
    const long long z   = idx / ((long long)DHd * NKd);
    const long long rem = idx - z * (long long)DHd * NKd;
    const int d = (int)(rem / NKd);
    const int j = (int)(rem - (long long)d * NKd);
    const int b = (int)(z / nH);
    const int h = (int)(z - (long long)b * nH);
    Vt[idx] = V[(long long)b * sVb + (long long)j * ldv + h * DHd + d];
  }
}

// LayerNorm over last dim D (D multiple of 32, D<=1024), one wave32 per row,
// emits bf16 for consumption by the next WMMA GEMM.
__global__ __launch_bounds__(256) void ln_to_bf16_kernel(
    const float* __restrict__ X, const float* __restrict__ w,
    const float* __restrict__ bia, unsigned short* __restrict__ Y,
    int D, float eps)
{
  const long long row = (long long)blockIdx.x * 8 + (threadIdx.x >> 5);
  const int lane = threadIdx.x & 31;
  const float* x = X + row * (long long)D;
  const int cnt = D >> 5;
  float v[32];
  float s = 0.f;
  for (int i = 0; i < cnt; ++i) { v[i] = x[lane + 32 * i]; s += v[i]; }
  s = wave_sum(s);
  const float mean = s / (float)D;
  float q = 0.f;
  for (int i = 0; i < cnt; ++i) { const float d = v[i] - mean; q += d * d; }
  q = wave_sum(q);
  const float rstd = rsqrtf(q / (float)D + eps);
  unsigned short* y = Y + row * (long long)D;
  for (int i = 0; i < cnt; ++i) {
    const int c = lane + 32 * i;
    y[c] = f2bf((v[i] - mean) * rstd * w[c] + bia[c]);
  }
}

// Row softmax with fused pre-scale (1/sqrt(DH)), emits bf16 probabilities.
__global__ __launch_bounds__(256) void softmax_bf16_kernel(
    const float* __restrict__ S, unsigned short* __restrict__ P,
    int n, float scale)
{
  const long long row = (long long)blockIdx.x * 8 + (threadIdx.x >> 5);
  const int lane = threadIdx.x & 31;
  const float* s = S + row * (long long)n;
  const int cnt = n >> 5;
  float v[32];
  float mx = -3.402823466e+38f;
  for (int i = 0; i < cnt; ++i) { v[i] = scale * s[lane + 32 * i]; mx = fmaxf(mx, v[i]); }
  mx = wave_max(mx);
  float sum = 0.f;
  for (int i = 0; i < cnt; ++i) { v[i] = __expf(v[i] - mx); sum += v[i]; }
  sum = wave_sum(sum);
  const float inv = 1.f / sum;
  unsigned short* p = P + row * (long long)n;
  for (int i = 0; i < cnt; ++i) p[lane + 32 * i] = f2bf(v[i] * inv);
}

__global__ void f32_to_bf16_kernel(const float* __restrict__ x,
                                   unsigned short* __restrict__ y, long long n)
{
  long long i = (long long)blockIdx.x * blockDim.x + threadIdx.x;
  const long long stride = (long long)gridDim.x * blockDim.x;
  for (; i < n; i += stride) y[i] = f2bf(x[i]);
}

__global__ void gelu_to_bf16_kernel(const float* __restrict__ x,
                                    unsigned short* __restrict__ y, long long n)
{
  long long i = (long long)blockIdx.x * blockDim.x + threadIdx.x;
  const long long stride = (long long)gridDim.x * blockDim.x;
  for (; i < n; i += stride) {
    const float t = x[i];
    y[i] = f2bf(0.5f * t * (1.f + erff(t * 0.70710678118654752f)));
  }
}

// gate logits (2x dot over 512) + 2-way softmax + blend, one wave per row.
__global__ __launch_bounds__(256) void mix_blend_kernel(
    const float* __restrict__ gctx, const float* __restrict__ mix_w,
    const float* __restrict__ mix_b, const float* __restrict__ selfo,
    const float* __restrict__ crosso, float* __restrict__ mixed,
    unsigned short* __restrict__ mixed_bf, int D)
{
  const long long row = (long long)blockIdx.x * 8 + (threadIdx.x >> 5);
  const int lane = threadIdx.x & 31;
  const float* g = gctx + row * (long long)D;
  float d0 = 0.f, d1 = 0.f;
  for (int c = lane; c < D; c += 32) {
    const float gv = g[c];
    d0 += gv * mix_w[c];
    d1 += gv * mix_w[D + c];
  }
  d0 = wave_sum(d0) + mix_b[0];
  d1 = wave_sum(d1) + mix_b[1];
  const float m  = fmaxf(d0, d1);
  const float e0 = __expf(d0 - m), e1 = __expf(d1 - m);
  const float p0 = e0 / (e0 + e1), p1 = 1.f - p0;
  const float* so = selfo  + row * (long long)D;
  const float* co = crosso + row * (long long)D;
  float* mo = mixed + row * (long long)D;
  unsigned short* mb = mixed_bf + row * (long long)D;
  for (int c = lane; c < D; c += 32) {
    const float val = p0 * so[c] + p1 * co[c];
    mo[c] = val;
    mb[c] = f2bf(val);
  }
}

__global__ void residual_kernel(const float* __restrict__ delta,
                                const float* __restrict__ ff2,
                                const float* __restrict__ gate,
                                float* __restrict__ out, long long n)
{
  long long i = (long long)blockIdx.x * blockDim.x + threadIdx.x;
  const long long stride = (long long)gridDim.x * blockDim.x;
  const float g = gate[0];
  for (; i < n; i += stride) out[i] = delta[i] + g * ff2[i];
}

// ---------------------------------------------------------------------------

extern "C" void kernel_launch(void* const* d_in, const int* in_sizes, int n_in,
                              void* d_out, int out_size, void* d_ws, size_t ws_size,
                              hipStream_t stream)
{
  (void)in_sizes; (void)n_in; (void)out_size; (void)ws_size;

  constexpr int  Bb = 2, NQ = 1024, NK = 1024, H = 8, DH = 64;
  constexpr int  INNER = 512, OD = 1024, QD = 1024, KVD = 1024, FF = 4096;
  constexpr int  ROWS = Bb * NQ;                       // 2048 (also Bb*NK)
  constexpr long long SROWS = (long long)Bb * H * NQ;  // softmax rows

  const float* query     = (const float*)d_in[0];
  const float* kvf       = (const float*)d_in[1];
  const float* nq_w      = (const float*)d_in[2];
  const float* nq_b      = (const float*)d_in[3];
  const float* nkv_w     = (const float*)d_in[4];
  const float* nkv_b     = (const float*)d_in[5];
  const float* wq_cross  = (const float*)d_in[6];
  const float* wkv_cross = (const float*)d_in[7];
  const float* wqkv_self = (const float*)d_in[8];
  const float* gn_w      = (const float*)d_in[9];
  const float* gn_b      = (const float*)d_in[10];
  const float* mha_in_w  = (const float*)d_in[11];
  const float* mha_out_w = (const float*)d_in[12];
  const float* mix_w     = (const float*)d_in[13];
  const float* mix_b     = (const float*)d_in[14];
  const float* w_out     = (const float*)d_in[15];
  const float* ff_ln_w   = (const float*)d_in[16];
  const float* ff_ln_b   = (const float*)d_in[17];
  const float* ff_fc1    = (const float*)d_in[18];
  const float* ff_fc2    = (const float*)d_in[19];
  const float* ff_gate   = (const float*)d_in[20];

  float* out    = (float*)d_out;                       // delta  [B,NQ,OD]
  float* out_wt = out + (long long)Bb * NQ * OD;       // wt_out [B,NQ,OD]

  // ---- bump allocator over d_ws (~320 MB total) ----
  char*  base = (char*)d_ws;
  size_t off  = 0;
  auto aU = [&](long long n) -> unsigned short* {
    unsigned short* p = (unsigned short*)(base + off);
    off = (off + (size_t)n * 2 + 255) & ~(size_t)255;
    return p;
  };
  auto aF = [&](long long n) -> float* {
    float* p = (float*)(base + off);
    off = (off + (size_t)n * 4 + 255) & ~(size_t)255;
    return p;
  };

  // bf16 weights (converted once per launch)
  unsigned short* wbf_qkv  = aU((long long)3 * INNER * QD);     // 1536x1024
  unsigned short* wbf_kvc  = aU((long long)2 * INNER * KVD);    // 1024x1024
  unsigned short* wbf_qc   = aU((long long)INNER * QD);         //  512x1024
  unsigned short* wbf_mha  = aU((long long)3 * INNER * INNER);  // 1536x512
  unsigned short* wbf_mout = aU((long long)INNER * INNER);      //  512x512
  unsigned short* wbf_wout = aU((long long)OD * INNER);         // 1024x512
  unsigned short* wbf_fc1  = aU((long long)FF * OD);            // 4096x1024
  unsigned short* wbf_fc2  = aU((long long)OD * FF);            // 1024x4096

  // bf16 activations
  unsigned short* kvn_bf   = aU((long long)ROWS * KVD);
  unsigned short* qn_bf    = aU((long long)ROWS * QD);
  unsigned short* qkvwt_bf = aU((long long)ROWS * 3 * INNER);
  unsigned short* qkvs_bf  = aU((long long)ROWS * 3 * INNER);
  unsigned short* qc_bf    = aU((long long)ROWS * INNER);
  unsigned short* kvc_bf   = aU((long long)ROWS * 2 * INNER);
  unsigned short* P_bf     = aU(SROWS * NK);                    // probs (reused x4)
  unsigned short* Vt_bf    = aU((long long)Bb * H * DH * NK);   // V^T (reused x4)
  unsigned short* son_bf   = aU((long long)ROWS * INNER);
  unsigned short* con_bf   = aU((long long)ROWS * INNER);
  unsigned short* gq_bf    = aU((long long)ROWS * INNER);
  unsigned short* gkv_bf   = aU((long long)ROWS * 2 * INNER);
  unsigned short* gao_bf   = aU((long long)ROWS * INNER);
  unsigned short* wtatt_bf = aU((long long)ROWS * INNER);
  unsigned short* mixed_bf = aU((long long)ROWS * INNER);
  unsigned short* ffln_bf  = aU((long long)ROWS * OD);
  unsigned short* gelu_bf  = aU((long long)ROWS * FF);

  // f32 activations
  float* qkv_wt = aF((long long)ROWS * 3 * INNER);
  float* qkv_s  = aF((long long)ROWS * 3 * INNER);
  float* q_c    = aF((long long)ROWS * INNER);
  float* kv_c   = aF((long long)ROWS * 2 * INNER);
  float* Smat   = aF(SROWS * NK);                               // scores (reused x4)
  float* selfo  = aF((long long)ROWS * INNER);
  float* crosso = aF((long long)ROWS * INNER);
  float* wtatt  = aF((long long)ROWS * INNER);
  float* gq     = aF((long long)ROWS * INNER);
  float* gkv    = aF((long long)ROWS * 2 * INNER);
  float* gao    = aF((long long)ROWS * INNER);
  float* gctx   = aF((long long)ROWS * INNER);
  float* mixed  = aF((long long)ROWS * INNER);
  float* delta  = aF((long long)ROWS * OD);
  float* ff1    = aF((long long)ROWS * FF);
  float* ff2    = aF((long long)ROWS * OD);

  // ---- launch helpers ----
  auto conv = [&](const float* x, unsigned short* y, long long n) {
    f32_to_bf16_kernel<<<dim3(1024), dim3(256), 0, stream>>>(x, y, n);
  };
  auto gemmNT = [&](const unsigned short* A, int lda, long long sAb, long long sAh,
                    const unsigned short* Bm, int ldb, long long sBb, long long sBh,
                    float* C, int ldc, long long sCb, long long sCh,
                    int M, int N, int K, int Z, int nH) {
    dim3 grid((N + 127) / 128, (M + 127) / 128, Z);
    gemm_bf16_wmma<<<grid, 256, 0, stream>>>(A, lda, sAb, sAh, Bm, ldb, sBb,
                                             sBh, C, ldc, sCb, sCh, M, N, K, nH);
  };
  auto ln = [&](const float* X, const float* w, const float* b,
                unsigned short* Y, int D, int nrows) {
    ln_to_bf16_kernel<<<dim3(nrows / 8), 256, 0, stream>>>(X, w, b, Y, D, 1e-5f);
  };
  // one full multi-head attention:  O[rows,512] = softmax(Q K^T/8) V  per head
  auto attention = [&](const unsigned short* Qbf, int ldq,
                       const unsigned short* Kbf, int ldk,
                       const unsigned short* Vbf, int ldv, float* O) {
    gemmNT(Qbf, ldq, (long long)NQ * ldq, DH,
           Kbf, ldk, (long long)NK * ldk, DH,
           Smat, NK, (long long)H * NQ * NK, (long long)NQ * NK,
           NQ, NK, DH, Bb * H, H);
    softmax_bf16_kernel<<<dim3((unsigned)(SROWS / 8)), 256, 0, stream>>>(
        Smat, P_bf, NK, 0.125f /* 1/sqrt(64) */);
    vtrans_kernel<<<dim3(1024), dim3(256), 0, stream>>>(
        Vbf, ldv, (long long)NK * ldv, H, NK, DH, Vt_bf,
        (long long)Bb * H * DH * NK);
    gemmNT(P_bf, NK, (long long)H * NQ * NK, (long long)NQ * NK,
           Vt_bf, NK, (long long)H * DH * NK, (long long)DH * NK,
           O, INNER, (long long)NQ * INNER, DH,
           NQ, DH, NK, Bb * H, H);
  };

  // ---- 0. weights -> bf16 ----
  conv(wqkv_self, wbf_qkv,  (long long)3 * INNER * QD);
  conv(wkv_cross, wbf_kvc,  (long long)2 * INNER * KVD);
  conv(wq_cross,  wbf_qc,   (long long)INNER * QD);
  conv(mha_in_w,  wbf_mha,  (long long)3 * INNER * INNER);
  conv(mha_out_w, wbf_mout, (long long)INNER * INNER);
  conv(w_out,     wbf_wout, (long long)OD * INNER);
  conv(ff_fc1,    wbf_fc1,  (long long)FF * OD);
  conv(ff_fc2,    wbf_fc2,  (long long)OD * FF);

  // ---- 1. input layernorms ----
  ln(kvf,   nkv_w, nkv_b, kvn_bf, KVD, ROWS);
  ln(query, nq_w,  nq_b,  qn_bf,  QD,  ROWS);

  // ---- 2. wt self-attention branch ----
  gemmNT(kvn_bf, KVD, 0, 0, wbf_qkv, QD, 0, 0, qkv_wt, 3 * INNER, 0, 0,
         ROWS, 3 * INNER, KVD, 1, 1);
  conv(qkv_wt, qkvwt_bf, (long long)ROWS * 3 * INNER);
  attention(qkvwt_bf,             3 * INNER,   // q_wt  (cols   0..511)
            qkvwt_bf + INNER,     3 * INNER,   // k_wt  (cols 512..1023)
            qkvwt_bf + 2 * INNER, 3 * INNER,   // v_wt  (cols 1024..1535)
            wtatt);
  conv(wtatt, wtatt_bf, (long long)ROWS * INNER);
  gemmNT(wtatt_bf, INNER, 0, 0, wbf_wout, INNER, 0, 0, out_wt, OD, 0, 0,
         ROWS, OD, INNER, 1, 1);               // wt_out -> d_out[1]

  // ---- 3. cross attention ----
  gemmNT(qn_bf, QD, 0, 0, wbf_qc, QD, 0, 0, q_c, INNER, 0, 0,
         ROWS, INNER, QD, 1, 1);
  conv(q_c, qc_bf, (long long)ROWS * INNER);
  gemmNT(kvn_bf, KVD, 0, 0, wbf_kvc, KVD, 0, 0, kv_c, 2 * INNER, 0, 0,
         ROWS, 2 * INNER, KVD, 1, 1);
  conv(kv_c, kvc_bf, (long long)ROWS * 2 * INNER);
  attention(qc_bf, INNER, kvc_bf, 2 * INNER, kvc_bf + INNER, 2 * INNER, crosso);

  // ---- 4. self attention on query ----
  gemmNT(qn_bf, QD, 0, 0, wbf_qkv, QD, 0, 0, qkv_s, 3 * INNER, 0, 0,
         ROWS, 3 * INNER, QD, 1, 1);
  conv(qkv_s, qkvs_bf, (long long)ROWS * 3 * INNER);
  attention(qkvs_bf, 3 * INNER, qkvs_bf + INNER, 3 * INNER,
            qkvs_bf + 2 * INNER, 3 * INNER, selfo);

  // ---- 5. gating MHA ----
  ln(selfo,  gn_w, gn_b, son_bf, INNER, ROWS);
  ln(crosso, gn_w, gn_b, con_bf, INNER, ROWS);
  gemmNT(son_bf, INNER, 0, 0, wbf_mha, INNER, 0, 0, gq, INNER, 0, 0,
         ROWS, INNER, INNER, 1, 1);                         // q = son @ wq_g^T
  conv(gq, gq_bf, (long long)ROWS * INNER);
  gemmNT(con_bf, INNER, 0, 0, wbf_mha + (long long)INNER * INNER, INNER, 0, 0,
         gkv, 2 * INNER, 0, 0, ROWS, 2 * INNER, INNER, 1, 1); // k|v = con @ [wk;wv]^T
  conv(gkv, gkv_bf, (long long)ROWS * 2 * INNER);
  attention(gq_bf, INNER, gkv_bf, 2 * INNER, gkv_bf + INNER, 2 * INNER, gao);
  conv(gao, gao_bf, (long long)ROWS * INNER);
  gemmNT(gao_bf, INNER, 0, 0, wbf_mout, INNER, 0, 0, gctx, INNER, 0, 0,
         ROWS, INNER, INNER, 1, 1);

  // ---- 6. mix + delta ----
  mix_blend_kernel<<<dim3(ROWS / 8), 256, 0, stream>>>(
      gctx, mix_w, mix_b, selfo, crosso, mixed, mixed_bf, INNER);
  gemmNT(mixed_bf, INNER, 0, 0, wbf_wout, INNER, 0, 0, delta, OD, 0, 0,
         ROWS, OD, INNER, 1, 1);

  // ---- 7. gated feed-forward ----
  ln(delta, ff_ln_w, ff_ln_b, ffln_bf, OD, ROWS);
  gemmNT(ffln_bf, OD, 0, 0, wbf_fc1, OD, 0, 0, ff1, FF, 0, 0,
         ROWS, FF, OD, 1, 1);
  gelu_to_bf16_kernel<<<dim3(2048), dim3(256), 0, stream>>>(
      ff1, gelu_bf, (long long)ROWS * FF);
  gemmNT(gelu_bf, FF, 0, 0, wbf_fc2, FF, 0, 0, ff2, OD, 0, 0,
         ROWS, OD, FF, 1, 1);
  residual_kernel<<<dim3(1024), dim3(256), 0, stream>>>(
      delta, ff2, ff_gate, out, (long long)ROWS * OD);      // delta -> d_out[0]
}
